// GCNEncoder_18915035972100
// MI455X (gfx1250) — compile-verified, matrix operand order checked by reading, not run on hardware
//
#include <hip/hip_runtime.h>

typedef __attribute__((ext_vector_type(16))) _Float16 v16h;
typedef __attribute__((ext_vector_type(8)))  _Float16 v8h;
typedef __attribute__((ext_vector_type(8)))  float    v8f;

// ---------------- utility kernels ----------------

__global__ void k_zero_f32(float* __restrict__ p, long n) {
  long i = (long)blockIdx.x * blockDim.x + threadIdx.x;
  if (i < n) p[i] = 0.0f;
}

__global__ void k_cvt_f16(const float* __restrict__ in, _Float16* __restrict__ out, long n) {
  long i = (long)blockIdx.x * blockDim.x + threadIdx.x;
  if (i < n) out[i] = (_Float16)in[i];
}

// W[K][C] row-major -> WT[C][K] (f16), so B columns are contiguous in K
__global__ void k_transpose_f16(const float* __restrict__ W, _Float16* __restrict__ WT,
                                int K, int C) {
  int i = blockIdx.x * blockDim.x + threadIdx.x;
  if (i < K * C) {
    int k = i / C, c = i % C;
    WT[(long)c * K + k] = (_Float16)W[i];
  }
}

// deg[dst] += 1 per edge
__global__ void k_deg_edges(const long long* __restrict__ ei, long E, float* __restrict__ deg) {
  long e = (long)blockIdx.x * blockDim.x + threadIdx.x;
  if (e < E) {
    long d = ei[E + e];
    atomicAdd(&deg[d], 1.0f);
  }
}

// dinv = rsqrt(deg + 1)   (self-loop weight folded in)
__global__ void k_finish_dinv(float* __restrict__ deg, long n) {
  long i = (long)blockIdx.x * blockDim.x + threadIdx.x;
  if (i < n) deg[i] = rsqrtf(deg[i] + 1.0f);
}

// ---------------- WMMA GEMM: C[M,NC] = A[M,K](f16) * B[K,NC] via BT[NC,K](f16) ----------------
// one wave computes one 16x16 output tile; K-loop of v_wmma_f32_16x16x32_f16

__global__ __launch_bounds__(256) void k_gemm_f16_wmma(
    const _Float16* __restrict__ A, const _Float16* __restrict__ BT,
    float* __restrict__ Cout, int M, int K, int NC) {
  int wave = threadIdx.x >> 5;
  int lane = threadIdx.x & 31;
  int tileM = blockIdx.x * 8 + wave;
  int tileN = blockIdx.y;
  int Mtiles = (M + 15) >> 4;
  if (tileM >= Mtiles) return;          // wave-uniform: EXEC stays all-1 for WMMA

  int half = lane >> 4;                 // 0: K 0-7 / 16-23, 1: K 8-15 / 24-31
  int m    = lane & 15;
  int row  = tileM * 16 + m;  if (row >= M) row = M - 1;   // clamp (duplicate loads ok)
  int col  = tileN * 16 + m;                                // B column per lane

  v8f acc = {};
  for (int kb = 0; kb < K; kb += 32) {
    const _Float16* ap = A  + (long)row * K + kb + half * 8;
    const _Float16* bp = BT + (long)col * K + kb + half * 8;
    v8h alo = *(const v8h*)(ap);
    v8h ahi = *(const v8h*)(ap + 16);
    v8h blo = *(const v8h*)(bp);
    v8h bhi = *(const v8h*)(bp + 16);
    v16h a, b;
#pragma unroll
    for (int i = 0; i < 8; i++) { a[i] = alo[i]; a[8 + i] = ahi[i]; b[i] = blo[i]; b[8 + i] = bhi[i]; }
    acc = __builtin_amdgcn_wmma_f32_16x16x32_f16(false, a, false, b, (short)0, acc, false, false);
  }

  // C layout: lanes 0-15 -> M = r, lanes 16-31 -> M = 8 + r; N = lane&15
  int mbase = tileM * 16 + half * 8;
  int n     = tileN * 16 + (lane & 15);
#pragma unroll
  for (int r = 0; r < 8; r++) {
    int rr = mbase + r;
    if (rr < M) Cout[(long)rr * NC + n] = acc[r];
  }
}

// ---------------- edge scatter: out[dst,:] += dinv[src]*dinv[dst] * h[src,:] ----------------
// one wave per edge; each lane owns C/32 contiguous features

__global__ void k_scatter_edges(const long long* __restrict__ ei, long E,
                                const float* __restrict__ dinv,
                                const float* __restrict__ h,
                                float* __restrict__ out, int C) {
  long gid = (long)blockIdx.x * blockDim.x + threadIdx.x;
  long e = gid >> 5;
  int lane = (int)(gid & 31);
  if (e >= E) return;
  long s = ei[e];
  long d = ei[E + e];
  float norm = dinv[s] * dinv[d];
  int fpl = C >> 5;
  const float* hp = h   + s * (long)C + lane * fpl;
  float*       op = out + d * (long)C + lane * fpl;
#pragma unroll 4
  for (int j = 0; j < fpl; j++) atomicAdd(&op[j], norm * hp[j]);
}

// a1h = f16( relu( agg + dinv^2 * h + b ) )
__global__ void k_selfloop_relu_f16(const float* __restrict__ agg, const float* __restrict__ h,
                                    const float* __restrict__ dinv, const float* __restrict__ b,
                                    _Float16* __restrict__ a_out, long N, int C) {
  long i = (long)blockIdx.x * blockDim.x + threadIdx.x;
  if (i >= N * (long)C) return;
  long node = i / C;
  int  c    = (int)(i - node * C);
  float di = dinv[node];
  float v  = agg[i] + di * di * h[i] + b[c];
  a_out[i] = (_Float16)(v > 0.0f ? v : 0.0f);
}

// out += dinv^2 * h + b   (final layer epilogue, no relu)
__global__ void k_selfloop_bias(float* __restrict__ out, const float* __restrict__ h,
                                const float* __restrict__ dinv, const float* __restrict__ b,
                                long N, int C) {
  long i = (long)blockIdx.x * blockDim.x + threadIdx.x;
  if (i >= N * (long)C) return;
  long node = i / C;
  int  c    = (int)(i - node * C);
  float di = dinv[node];
  out[i] += di * di * h[i] + b[c];
}

// ---------------- host launcher ----------------

static inline size_t align256(size_t v) { return (v + 255) & ~(size_t)255; }

extern "C" void kernel_launch(void* const* d_in, const int* in_sizes, int n_in,
                              void* d_out, int out_size, void* d_ws, size_t ws_size,
                              hipStream_t stream) {
  const float*     x  = (const float*)d_in[0];
  const long long* ei = (const long long*)d_in[1];
  const float*     W1 = (const float*)d_in[2];
  const float*     b1 = (const float*)d_in[3];
  const float*     W2 = (const float*)d_in[4];
  const float*     b2 = (const float*)d_in[5];

  const int  dh   = in_sizes[3];            // 128
  const int  dout = in_sizes[5];            // 64
  const int  din  = in_sizes[2] / dh;       // 128
  const long N    = (long)in_sizes[0] / din;
  const long E    = (long)in_sizes[1] / 2;

  // workspace carve
  uint8_t* base = (uint8_t*)d_ws;
  size_t off = 0;
  float*    dinv = (float*)(base + off);    off += align256((size_t)N * 4);
  _Float16* Xh   = (_Float16*)(base + off); off += align256((size_t)N * din * 2);
  _Float16* W1T  = (_Float16*)(base + off); off += align256((size_t)din * dh * 2);
  _Float16* W2T  = (_Float16*)(base + off); off += align256((size_t)dh * dout * 2);
  float*    h1   = (float*)(base + off);    off += align256((size_t)N * dh * 4);
  float*    agg1 = (float*)(base + off);    off += align256((size_t)N * dh * 4);
  _Float16* a1h  = Xh;    // reuse: Xh dead after GEMM1 (stream-ordered)
  float*    h2   = agg1;  // reuse: agg1 dead after selfloop_relu (stream-ordered)
  float*    out  = (float*)d_out;

  const int BT_ = 256;
  auto blocks = [&](long n) { return (unsigned)((n + BT_ - 1) / BT_); };

  // --- degree / dinv ---
  k_zero_f32<<<blocks(N), BT_, 0, stream>>>(dinv, N);
  k_deg_edges<<<blocks(E), BT_, 0, stream>>>(ei, E, dinv);
  k_finish_dinv<<<blocks(N), BT_, 0, stream>>>(dinv, N);

  // --- f16 conversions ---
  k_cvt_f16<<<blocks(N * din), BT_, 0, stream>>>(x, Xh, N * din);
  k_transpose_f16<<<blocks((long)din * dh), BT_, 0, stream>>>(W1, W1T, din, dh);
  k_transpose_f16<<<blocks((long)dh * dout), BT_, 0, stream>>>(W2, W2T, dh, dout);

  // --- layer 1: h1 = Xh @ W1 (WMMA) ---
  {
    int Mtiles = (int)((N + 15) / 16);
    dim3 grid((Mtiles + 7) / 8, dh / 16);
    k_gemm_f16_wmma<<<grid, 256, 0, stream>>>(Xh, W1T, h1, (int)N, din, dh);
  }
  k_zero_f32<<<blocks(N * dh), BT_, 0, stream>>>(agg1, N * dh);
  k_scatter_edges<<<blocks(E * 32), BT_, 0, stream>>>(ei, E, dinv, h1, agg1, dh);
  k_selfloop_relu_f16<<<blocks(N * dh), BT_, 0, stream>>>(agg1, h1, dinv, b1, a1h, N, dh);

  // --- layer 2: h2 = a1h @ W2 (WMMA) ---
  {
    int Mtiles = (int)((N + 15) / 16);
    dim3 grid((Mtiles + 7) / 8, dout / 16);
    k_gemm_f16_wmma<<<grid, 256, 0, stream>>>(a1h, W2T, h2, (int)N, dh, dout);
  }
  k_zero_f32<<<blocks(N * dout), BT_, 0, stream>>>(out, N * dout);
  k_scatter_edges<<<blocks(E * 32), BT_, 0, stream>>>(ei, E, dinv, h2, out, dout);
  k_selfloop_bias<<<blocks(N * dout), BT_, 0, stream>>>(out, h2, dinv, b2, N, dout);
}
